// SeqAttnMatch_28200755265639
// MI455X (gfx1250) — compile-verified, matrix operand order checked by reading, not run on hardware
//
#include <hip/hip_runtime.h>

// ---------------------------------------------------------------------------
// SeqAttnMatch fused attention for gfx1250 (MI455X), wave32 + WMMA f16 + TDM.
//   B=32, LP=2048, LQ=512, D=256
// ---------------------------------------------------------------------------

typedef _Float16 v16h __attribute__((ext_vector_type(16)));
typedef _Float16 v8h  __attribute__((ext_vector_type(8)));
typedef float    v8f  __attribute__((ext_vector_type(8)));
typedef float    v4f  __attribute__((ext_vector_type(4)));

typedef unsigned int u32x4 __attribute__((ext_vector_type(4)));
typedef int          i32x4 __attribute__((ext_vector_type(4)));
typedef int          i32x8 __attribute__((ext_vector_type(8)));

union AB16 { v16h v; v8h h[2]; };

// WMMA f16 -> f32 accum, 16x16x32.
__device__ __forceinline__ v8f wmma16(v16h a, v16h b, v8f c) {
    return __builtin_amdgcn_wmma_f32_16x16x32_f16(
        /*neg_a=*/false, a, /*neg_b=*/false, b,
        /*c_mod=*/(short)0, c, /*reuse_a=*/false, /*reuse_b=*/false);
}

// Load one 16x32 A-chunk from an f32 row-major source (row stride 256),
// converting to f16 per the CDNA5 16-bit A layout:
//   lane = g*16 + m : row M=m ; halves 0..7 = K = k0+g*8+0..7,
//                               halves 8..15 = K = k0+16+g*8+0..7
__device__ __forceinline__ v16h load_a_f32(const float* rowp, int k0, int g) {
    const v4f* p0 = (const v4f*)(rowp + k0 + g * 8);
    const v4f* p1 = (const v4f*)(rowp + k0 + 16 + g * 8);
    v4f f0 = p0[0], f1 = p0[1], f2 = p1[0], f3 = p1[1];
    v16h a;
#pragma unroll
    for (int i = 0; i < 4; ++i) {
        a[i]      = (_Float16)f0[i];
        a[4 + i]  = (_Float16)f1[i];
        a[8 + i]  = (_Float16)f2[i];
        a[12 + i] = (_Float16)f3[i];
    }
    return a;
}

// Load one 32x16 B-chunk where lane's column maps to a row-major f16 row
// (NT-gemm pattern): two contiguous 16-byte loads per lane.
__device__ __forceinline__ v16h load_b_f16(const _Float16* rowp, int k0, int g) {
    AB16 u;
    u.h[0] = *(const v8h*)(rowp + k0 + g * 8);
    u.h[1] = *(const v8h*)(rowp + k0 + 16 + g * 8);
    return u.v;
}

// Generic-pointer low 32 bits == LDS byte offset (flat aperture mapping).
__device__ __forceinline__ unsigned lds_off(const void* p) {
    return (unsigned)(size_t)p;
}

// TDM: async DMA of a 16-row x 256-half (8 KB) f16 tile, row stride 256
// halves, from global memory into LDS.  Tracked by TENSORcnt.
//   D# group0: count=1 | lds_addr | global_addr | type=2
//   D# group1: data_size=1(2B), tensor_dim0=256, tensor_dim1=16,
//              tile_dim0=256, tile_dim1=16, tensor_dim0_stride=256
__device__ __forceinline__ void tdm_load_16x256_f16(const _Float16* gptr,
                                                    unsigned lds_addr) {
    unsigned long long ga = (unsigned long long)(size_t)gptr;
    u32x4 g0;
    g0[0] = 1u;                                   // count=1, user descriptor
    g0[1] = lds_addr;                             // bits [63:32]
    g0[2] = (unsigned)ga;                         // global_addr[31:0]
    g0[3] = (unsigned)(ga >> 32) | (2u << 30);    // global_addr[56:32] | type=2
    i32x8 g1;
    g1[0] = (int)(1u << 16);                      // data_size = 1 -> 2 bytes
    g1[1] = (int)(256u << 16);                    // tensor_dim0[15:0] @ [63:48]
    g1[2] = (int)(16u << 16);                     // dim0 hi=0 | tensor_dim1=16
    g1[3] = (int)(256u << 16);                    // dim1 hi=0 | tile_dim0=256
    g1[4] = (int)16;                              // tile_dim1=16 | tile_dim2=0
    g1[5] = (int)256;                             // tensor_dim0_stride lo32
    g1[6] = 0;                                    // stride hi | dim1_stride lo
    g1[7] = 0;
    i32x4 z4 = {0, 0, 0, 0};
    i32x8 z8 = {0, 0, 0, 0, 0, 0, 0, 0};
    // 6-arg form (clang-23 / therock-10.0 headers).
    __builtin_amdgcn_tensor_load_to_lds(g0, g1, z4, z4, z8, 0);
}

// ---------------------------------------------------------------------------
// Kernel 1: cast W [256,256] f32 -> f16 row-major
// ---------------------------------------------------------------------------
__global__ void k_cast_w(const float* __restrict__ W, _Float16* __restrict__ W16) {
    int i = blockIdx.x * blockDim.x + threadIdx.x;  // 65536 total
    W16[i] = (_Float16)W[i];
}

// ---------------------------------------------------------------------------
// Kernel 2: yT16[b][d][q] = (f16) y[b][q][d]   (B=32, LQ=512, D=256)
// ---------------------------------------------------------------------------
__global__ void k_transpose_y(const float* __restrict__ y, _Float16* __restrict__ yT16) {
    int i = blockIdx.x * blockDim.x + threadIdx.x;  // 32*512*256 total
    int d = i & 255;
    int q = (i >> 8) & 511;
    int b = i >> 17;
    yT16[((size_t)b * 256 + d) * 512 + q] = (_Float16)y[i];
}

// ---------------------------------------------------------------------------
// Kernel 3: proj = relu(src @ W16^T + bias), f16 out.  One wave per 16 rows.
// ---------------------------------------------------------------------------
__global__ void k_proj(const float* __restrict__ src, const _Float16* __restrict__ W16,
                       const float* __restrict__ bias, _Float16* __restrict__ dst,
                       int nrows) {
    int wave = (blockIdx.x * blockDim.x + threadIdx.x) >> 5;
    int lane = threadIdx.x & 31;
    int m = lane & 15, g = lane >> 4;
    int row0 = wave * 16;
    if (row0 >= nrows) return;

    const float* rowp = src + (size_t)(row0 + m) * 256;
    v16h A[8];
#pragma unroll
    for (int ks = 0; ks < 8; ++ks) A[ks] = load_a_f32(rowp, ks * 32, g);

    for (int nc = 0; nc < 16; ++nc) {
        int e0 = nc * 16;
        float bv = bias[e0 + m];
        v8f acc;
#pragma unroll
        for (int r = 0; r < 8; ++r) acc[r] = bv;

        const _Float16* wrow = W16 + (size_t)(e0 + m) * 256;
#pragma unroll
        for (int ks = 0; ks < 8; ++ks)
            acc = wmma16(A[ks], load_b_f16(wrow, ks * 32, g), acc);

#pragma unroll
        for (int r = 0; r < 8; ++r) {
            int M = r + 8 * g;
            dst[(size_t)(row0 + M) * 256 + e0 + m] = (_Float16)fmaxf(acc[r], 0.f);
        }
    }
}

// ---------------------------------------------------------------------------
// Kernel 4: fused flash attention.  1 block = 4 waves = 16 LP rows of 1 batch.
//   grid = 32 * 128 blocks, 128 threads.
// ---------------------------------------------------------------------------
__global__ __launch_bounds__(128) void k_attn(
    const float* __restrict__ x, const _Float16* __restrict__ W16,
    const float* __restrict__ bias, const _Float16* __restrict__ yproj,
    const _Float16* __restrict__ yT16, float* __restrict__ out) {

    __shared__ _Float16 xp[16 * 256];          // x_proj tile (f16)   8 KB
    __shared__ float    S[16 * 512];           // scores (f32)       32 KB
    // 64 KB pool: stage B = per-wave TDM double buffers (4 waves x 2 x 8 KB)
    //             stage C/D = probability matrix P (16 x 512 f16, 16 KB)
    __shared__ __align__(16) _Float16 pool[4 * 2 * 16 * 256];
    __shared__ float    red[16 * 8];
    __shared__ float    red2[16 * 8];
    __shared__ float    linv[16];

    int b  = blockIdx.x >> 7;
    int p0 = (blockIdx.x & 127) * 16;
    int tid = threadIdx.x;
    int wave = tid >> 5, lane = tid & 31;
    int m = lane & 15, g = lane >> 4;

    // ---- Stage A: x_proj tile = relu(x_tile @ W^T + b) -> LDS (f16) ----
    const float* xrow = x + ((size_t)b * 2048 + p0 + m) * 256;
    v16h A[8];
#pragma unroll
    for (int ks = 0; ks < 8; ++ks) A[ks] = load_a_f32(xrow, ks * 32, g);

#pragma unroll
    for (int nc = 0; nc < 4; ++nc) {          // wave owns 64 output cols
        int e0 = wave * 64 + nc * 16;
        float bv = bias[e0 + m];
        v8f acc;
#pragma unroll
        for (int r = 0; r < 8; ++r) acc[r] = bv;
        const _Float16* wrow = W16 + (size_t)(e0 + m) * 256;
#pragma unroll
        for (int ks = 0; ks < 8; ++ks)
            acc = wmma16(A[ks], load_b_f16(wrow, ks * 32, g), acc);
#pragma unroll
        for (int r = 0; r < 8; ++r) {
            int M = r + 8 * g;
            xp[M * 256 + e0 + m] = (_Float16)fmaxf(acc[r], 0.f);
        }
    }
    __syncthreads();

    // ---- Stage B: scores = x_proj @ y_proj^T, B-operand via TDM -> LDS ----
    v16h XA[8];
#pragma unroll
    for (int ks = 0; ks < 8; ++ks)
        XA[ks] = load_b_f16(xp + m * 256, ks * 32, g);

    _Float16* buf0 = pool + wave * 2 * 4096;
    _Float16* buf1 = buf0 + 4096;
    const _Float16* ypbase = yproj + ((size_t)b * 512 + wave * 128) * 256;

    tdm_load_16x256_f16(ypbase, lds_off(buf0));            // prime buffer 0
#pragma unroll
    for (int qc = 0; qc < 8; ++qc) {
        if (qc + 1 < 8) {
            tdm_load_16x256_f16(ypbase + (size_t)(qc + 1) * 16 * 256,
                                lds_off((qc & 1) ? buf0 : buf1));
            __builtin_amdgcn_s_wait_tensorcnt(1);          // current tile ready
        } else {
            __builtin_amdgcn_s_wait_tensorcnt(0);
        }
        const _Float16* bb = (qc & 1) ? buf1 : buf0;
        v8f acc;
#pragma unroll
        for (int r = 0; r < 8; ++r) acc[r] = 0.f;
#pragma unroll
        for (int ks = 0; ks < 8; ++ks)
            acc = wmma16(XA[ks], load_b_f16(bb + m * 256, ks * 32, g), acc);
        int q0 = wave * 128 + qc * 16;
#pragma unroll
        for (int r = 0; r < 8; ++r)
            S[(r + 8 * g) * 512 + q0 + m] = acc[r];
    }
    __syncthreads();

    // ---- Stage C: softmax over LQ=512, write f16 probs (P overlays pool) ----
    _Float16* P = pool;
    {
        int row = tid >> 3, sl = tid & 7;     // 8 threads per row, 64 elems each
        const float* srow = S + row * 512 + sl * 64;
        float mx = -3.0e38f;
        for (int i = 0; i < 64; ++i) mx = fmaxf(mx, srow[i]);
        red[row * 8 + sl] = mx;
        __syncthreads();
        float rmax = red[row * 8];
        for (int i = 1; i < 8; ++i) rmax = fmaxf(rmax, red[row * 8 + i]);
        _Float16* prow = P + row * 512 + sl * 64;
        float s = 0.f;
        for (int i = 0; i < 64; ++i) {
            float e = __expf(srow[i] - rmax);
            prow[i] = (_Float16)e;
            s += e;
        }
        red2[row * 8 + sl] = s;
        __syncthreads();
        if (sl == 0) {
            float t = 0.f;
            for (int i = 0; i < 8; ++i) t += red2[row * 8 + i];
            linv[row] = 1.f / t;
        }
        __syncthreads();
    }

    // ---- Stage D: out = (P @ y) * (1/l), y via yT16 [b][d][q] ----
#pragma unroll
    for (int dc = 0; dc < 4; ++dc) {          // wave owns 64 d-columns
        int d0 = wave * 64 + dc * 16;
        const _Float16* ytrow = yT16 + ((size_t)b * 256 + d0 + m) * 512;
        __builtin_prefetch(ytrow, 0, 0);      // global_prefetch_b8
        v8f acc;
#pragma unroll
        for (int r = 0; r < 8; ++r) acc[r] = 0.f;
#pragma unroll
        for (int ks = 0; ks < 16; ++ks) {     // K = q, 512 total
            v16h PA = load_b_f16(P + m * 512, ks * 32, g);
            acc = wmma16(PA, load_b_f16(ytrow, ks * 32, g), acc);
        }
#pragma unroll
        for (int r = 0; r < 8; ++r) {
            int M = r + 8 * g;
            out[((size_t)b * 2048 + p0 + M) * 256 + d0 + m] = acc[r] * linv[M];
        }
    }
}

// ---------------------------------------------------------------------------
// Host-side launch
// ---------------------------------------------------------------------------
extern "C" void kernel_launch(void* const* d_in, const int* in_sizes, int n_in,
                              void* d_out, int out_size, void* d_ws, size_t ws_size,
                              hipStream_t stream) {
    (void)in_sizes; (void)n_in; (void)out_size; (void)ws_size;
    const float* x    = (const float*)d_in[0];   // [32,2048,256]
    const float* y    = (const float*)d_in[1];   // [32, 512,256]
    const float* W    = (const float*)d_in[2];   // [256,256]
    const float* bias = (const float*)d_in[3];   // [256]
    float* out = (float*)d_out;                  // [32,2048,256]

    char* ws = (char*)d_ws;
    _Float16* W16   = (_Float16*)ws;                             // 128 KB
    _Float16* yproj = (_Float16*)(ws + (1u << 17));              // 8 MB
    _Float16* yT16  = (_Float16*)(ws + (1u << 17) + (8u << 20)); // 8 MB

    k_cast_w     <<<256, 256, 0, stream>>>(W, W16);
    k_transpose_y<<<(32 * 512 * 256) / 256, 256, 0, stream>>>(y, yT16);
    k_proj       <<<128, 256, 0, stream>>>(y, W16, bias, yproj, 32 * 512);
    k_attn       <<<32 * 128, 128, 0, stream>>>(x, W16, bias, yproj, yT16, out);
}
